// MyConv2D_87522843558962
// MI455X (gfx1250) — matrix operand order repaired and checked by reading, not compiled, not run on hardware
//
#include <hip/hip_runtime.h>
#include <hip/hip_bf16.h>

// ---- vector types for WMMA ----
typedef __attribute__((ext_vector_type(16))) __bf16 v16bf;
typedef __attribute__((ext_vector_type(8)))  __bf16 v8bf;
typedef __attribute__((ext_vector_type(8)))  float  v8f;

// Conv: x (16,128,128,64) NHWC f32, kernel (3,3,64,128) HWIO f32, bias (128,)
// out (16,128,128,128) NHWC f32. Stride 1, SAME.
// Implicit GEMM: M = n*h*w (262144), N = cout (128), K = 3*3*64 (576).
#define CIN    64
#define COUT   128
#define WDIM   128
#define HDIM   128
#define KSTEPS 18   // 9 taps * 2 cin-halves, K=32 per step

// ------------------------------------------------------------------
// One-time weight repack: f32 HWIO -> bf16 [kstep][cout][32] in d_ws.
// ------------------------------------------------------------------
__global__ void conv_prep_weights(const float* __restrict__ kern,
                                  __bf16* __restrict__ wsB) {
    int idx = blockIdx.x * 256 + threadIdx.x;   // (s, kl, cout), cout innermost
    if (idx >= KSTEPS * 32 * COUT) return;
    int cout = idx & (COUT - 1);
    int kl   = (idx >> 7) & 31;
    int s    = idx >> 12;
    int tap  = s >> 1;
    int cin  = ((s & 1) * 32) + kl;
    float v = kern[((long)(tap * CIN + cin)) * COUT + cout];
    wsB[((long)(s * COUT + cout)) * 32 + kl] = (__bf16)v;
}

// ------------------------------------------------------------------
// Main kernel: one block = one (n,h) image row = 128x128 output tile.
// Double-buffered LDS; B tile via async global->LDS; one barrier/step.
// ------------------------------------------------------------------
__global__ void __launch_bounds__(256)
conv2d_wmma_bf16(const float* __restrict__ x,
                 const __bf16* __restrict__ wsB,
                 const float* __restrict__ bias,
                 float* __restrict__ out) {
    __shared__ __align__(32) __bf16 ldsA[2][128 * 32];   // [w][k]    2 x 8 KB
    __shared__ __align__(32) __bf16 ldsB[2][128 * 32];   // [cout][k] 2 x 8 KB

    const int tid    = threadIdx.x;
    const int lane   = tid & 31;
    const int wid    = tid >> 5;
    const int wm     = wid & 1;            // M wave offset: wm*64
    const int wn     = wid >> 1;           // N wave offset: wn*32
    const int lanelo = lane & 15;
    const int lanehi = lane >> 4;

    const int tileIdx = blockIdx.x;        // n*128 + h
    const int h       = tileIdx & (HDIM - 1);

    v8f acc[4][2];
#pragma unroll
    for (int mi = 0; mi < 4; ++mi)
#pragma unroll
        for (int ni = 0; ni < 2; ++ni)
            acc[mi][ni] = (v8f)(0.0f);

    // A-tile cooperative-load mapping: thread -> (w row, cin half)
    const int aw    = tid & 127;
    const int chalf = tid >> 7;

    // --- staged A registers (one K-step in flight) ---
    float4 ar0, ar1, ar2, ar3;
    bool   avalid = false;

    auto loadA = [&](int s) {
        const int tap = s >> 1;
        const int kh  = tap / 3;
        const int kw  = tap % 3;
        const int c0  = (s & 1) * 32;
        const int ih  = h + kh - 1;
        const int iw  = aw + kw - 1;
        avalid = ((unsigned)ih < (unsigned)HDIM) & ((unsigned)iw < (unsigned)WDIM);
        if (avalid) {
            const float4* src = (const float4*)(x +
                (((long)tileIdx - h + ih) * WDIM + iw) * CIN + c0 + chalf * 16);
            ar0 = src[0]; ar1 = src[1]; ar2 = src[2]; ar3 = src[3];
        }
    };

    auto storeA = [&](int s) {
        __bf16* dst = &ldsA[s & 1][aw * 32 + chalf * 16];
        v8bf o0, o1;
        if (avalid) {
            o0[0]=(__bf16)ar0.x; o0[1]=(__bf16)ar0.y; o0[2]=(__bf16)ar0.z; o0[3]=(__bf16)ar0.w;
            o0[4]=(__bf16)ar1.x; o0[5]=(__bf16)ar1.y; o0[6]=(__bf16)ar1.z; o0[7]=(__bf16)ar1.w;
            o1[0]=(__bf16)ar2.x; o1[1]=(__bf16)ar2.y; o1[2]=(__bf16)ar2.z; o1[3]=(__bf16)ar2.w;
            o1[4]=(__bf16)ar3.x; o1[5]=(__bf16)ar3.y; o1[6]=(__bf16)ar3.z; o1[7]=(__bf16)ar3.w;
        } else {
            o0 = (v8bf)((__bf16)0.0f);
            o1 = o0;
        }
        *(v8bf*)(dst)     = o0;
        *(v8bf*)(dst + 8) = o1;
    };

    // B tile: direct async global->LDS copy, 32 B per thread (2 x b128).
    // ISA 10.? async pseudocode: INST_OFFSET is added to BOTH the LDS and the
    // global address, so one (lds, gaddr) pair serves both halves.
    auto asyncB = [&](int s) {
        const __bf16* g = wsB + (long)s * COUT * 32 + tid * 16;
        unsigned ldsoff = (unsigned)(unsigned long long)&ldsB[s & 1][tid * 16];
        asm volatile(
            "global_load_async_to_lds_b128 %0, %1, off\n\t"
            "global_load_async_to_lds_b128 %0, %1, off offset:16"
            :: "v"(ldsoff), "v"(g) : "memory");
    };

    auto compute = [&](int s) {
        const __bf16* bA = ldsA[s & 1];
        const __bf16* bB = ldsB[s & 1];
        v16bf aF[4];
#pragma unroll
        for (int mi = 0; mi < 4; ++mi) {
            const int row = wm * 64 + mi * 16 + lanelo;
            const __bf16* p = bA + row * 32 + lanehi * 16;
            v8bf lo = *(const v8bf*)(p);          // K = 16*lanehi .. +7
            v8bf hi = *(const v8bf*)(p + 8);      // K = 16*lanehi+8 .. +7
            aF[mi] = __builtin_shufflevector(lo, hi,
                0,1,2,3,4,5,6,7,8,9,10,11,12,13,14,15);
        }
        v16bf bF[2];
#pragma unroll
        for (int ni = 0; ni < 2; ++ni) {
            const int row = wn * 32 + ni * 16 + lanelo;
            bF[ni] = *(const v16bf*)(bB + row * 32 + lanehi * 16);  // K = 16*lanehi..
        }
#pragma unroll
        for (int mi = 0; mi < 4; ++mi)
#pragma unroll
            for (int ni = 0; ni < 2; ++ni)
                acc[mi][ni] = __builtin_amdgcn_wmma_f32_16x16x32_bf16(
                    false, aF[mi], false, bF[ni],
                    (short)0, acc[mi][ni], false, false);
    };

    // ---- prologue: stage step 0 into buffer 0 ----
    asyncB(0);
    loadA(0);
    storeA(0);
    asm volatile("s_wait_asynccnt 0x0" ::: "memory");

    // ---- main loop: one barrier per K-step, fully unrolled ----
#pragma unroll
    for (int s = 0; s < KSTEPS; ++s) {
        __syncthreads();          // publishes buffer s&1; all waves past reads of (s+1)&1
        if (s + 1 < KSTEPS) {
            asyncB(s + 1);        // global->LDS, overlaps with compute below
            loadA(s + 1);         // global->VGPR, overlaps with compute below
        }
        compute(s);
        if (s + 1 < KSTEPS) {
            storeA(s + 1);        // ds_store into the other buffer
            asm volatile("s_wait_asynccnt 0x0" ::: "memory");  // B(s+1) resident pre-barrier
        }
    }

    // ---- epilogue: bias add + store ----
    float bv[2];
#pragma unroll
    for (int ni = 0; ni < 2; ++ni)
        bv[ni] = bias[wn * 32 + ni * 16 + lanelo];

#pragma unroll
    for (int mi = 0; mi < 4; ++mi) {
#pragma unroll
        for (int ni = 0; ni < 2; ++ni) {
            const int cout = wn * 32 + ni * 16 + lanelo;
#pragma unroll
            for (int r = 0; r < 8; ++r) {
                const int m = wm * 64 + mi * 16 + r + 8 * lanehi;
                out[((long)tileIdx * WDIM + m) * COUT + cout] = acc[mi][ni][r] + bv[ni];
            }
        }
    }
}

extern "C" void kernel_launch(void* const* d_in, const int* in_sizes, int n_in,
                              void* d_out, int out_size, void* d_ws, size_t ws_size,
                              hipStream_t stream) {
    const float* x    = (const float*)d_in[0];
    const float* kern = (const float*)d_in[1];
    const float* bias = (const float*)d_in[2];
    float* out        = (float*)d_out;
    __bf16* wsB       = (__bf16*)d_ws;   // needs 18*128*32*2 = 147,456 bytes

    conv_prep_weights<<<(KSTEPS * 32 * COUT + 255) / 256, 256, 0, stream>>>(kern, wsB);
    conv2d_wmma_bf16<<<16 * HDIM, 256, 0, stream>>>(x, wsB, bias, out);
}
